// AttentionModel_31078383354507
// MI455X (gfx1250) — matrix operand order repaired and checked by reading, not compiled, not run on hardware
//
#include <hip/hip_runtime.h>
#include <hip/hip_bf16.h>
#include <math.h>

typedef __bf16 bf16_t;
typedef bf16_t v16bf __attribute__((ext_vector_type(16)));
typedef bf16_t v8bf  __attribute__((ext_vector_type(8)));
typedef float  v8f   __attribute__((ext_vector_type(8)));

#define B_  32
#define T_  2048
#define F_  1024
#define D_  1024
#define LDS_STRIDE 1032   // padded halves per row: 2064B stride -> conflict-free ds_load_b128

// ---------------------------------------------------------------------------
// Branch-free tanh. gfx1250 has a native v_tanh_f32 (TRANS unit); fall back to
// a straight-line exp-based form (v_exp_f32 + fast div + copysign) otherwise.
// ---------------------------------------------------------------------------
#if defined(__has_builtin)
#if __has_builtin(__builtin_amdgcn_tanhf)
#define ATHENA_HAS_V_TANH 1
#endif
#endif

__device__ __forceinline__ float fast_tanh(float x) {
#ifdef ATHENA_HAS_V_TANH
    return __builtin_amdgcn_tanhf(x);
#else
    const float ax = fabsf(x);
    const float e  = __expf(-2.0f * ax);                 // v_exp_f32 path
    const float t  = __fdividef(1.0f - e, 1.0f + e);     // fast rcp-based div
    return copysignf(t, x);
#endif
}

// ---------------------------------------------------------------------------
// Kernel A: combinedBias[b,d] = sum_h hidden[b,h]*W2[h,d] + b1[d] + b2[d]
// ---------------------------------------------------------------------------
__global__ __launch_bounds__(256)
void prep_bias_kernel(const float* __restrict__ hidden,
                      const float* __restrict__ W2,
                      const float* __restrict__ b1,
                      const float* __restrict__ b2,
                      float* __restrict__ cb) {
    const int b   = blockIdx.x;
    const int tid = threadIdx.x;
    float acc[4];
#pragma unroll
    for (int j = 0; j < 4; ++j) acc[j] = 0.f;
    const float* hrow = hidden + b * D_;
    for (int h = 0; h < D_; ++h) {
        const float hv = hrow[h];               // uniform -> scalar load
        const float* wrow = W2 + (size_t)h * D_;
#pragma unroll
        for (int j = 0; j < 4; ++j)
            acc[j] = fmaf(hv, wrow[j * 256 + tid], acc[j]);
    }
#pragma unroll
    for (int j = 0; j < 4; ++j) {
        const int d = j * 256 + tid;
        cb[b * D_ + d] = acc[j] + b1[d] + b2[d];
    }
}

// ---------------------------------------------------------------------------
// Kernel A2: W1t[n,k] = bf16(W1[k,n])  (LDS-tiled 32x32 transpose + convert)
// ---------------------------------------------------------------------------
__global__ __launch_bounds__(256)
void transpose_w1_kernel(const float* __restrict__ W1, bf16_t* __restrict__ W1t) {
    __shared__ float tile[32][33];
    const int tx = threadIdx.x & 31;
    const int ty = threadIdx.x >> 5;
    const int n0 = (blockIdx.x & 31) * 32;
    const int k0 = (blockIdx.x >> 5) * 32;
#pragma unroll
    for (int j = 0; j < 4; ++j) {
        const int r = ty + j * 8;
        tile[r][tx] = W1[(size_t)(k0 + r) * D_ + (n0 + tx)];
    }
    __syncthreads();
#pragma unroll
    for (int j = 0; j < 4; ++j) {
        const int r = ty + j * 8;
        W1t[(size_t)(n0 + r) * F_ + (k0 + tx)] = (bf16_t)tile[tx][r];
    }
}

// ---------------------------------------------------------------------------
// Kernel B: fused  score[b,t] = bv + sum_n Wv[n]*tanh(feat@W1 + cb[b,n])
// Block = 16 rows of M=B*T staged once into LDS (bf16). 8 waves split the 64
// N-tiles, 2 N-tiles per K-sweep (A fragment reused by two WMMAs).
// ---------------------------------------------------------------------------
__global__ __launch_bounds__(256)
void attn_score_kernel(const float* __restrict__ features,
                       const bf16_t* __restrict__ W1t,
                       const float* __restrict__ cb,
                       const float* __restrict__ Wv,
                       const float* __restrict__ bv,
                       float* __restrict__ scores) {
    __shared__ __align__(16) bf16_t As[16 * LDS_STRIDE];
    __shared__ float partials[8][16];

    const int tid  = threadIdx.x;
    const int lane = tid & 31;
    const int wave = tid >> 5;
    const int m16  = blockIdx.x;          // 16-row tile index
    const int rowsBase = m16 * 16;
    const int b = rowsBase / T_;          // T_ % 16 == 0 -> whole tile same batch

    // ---- stage A tile: 16 rows x 1024 (fp32 -> bf16), coalesced float4 loads
    for (int r = 0; r < 16; ++r) {
        const float4 f4 = reinterpret_cast<const float4*>(
            features + (size_t)(rowsBase + r) * F_)[tid];
        bf16_t* dst = &As[r * LDS_STRIDE + tid * 4];
        dst[0] = (bf16_t)f4.x; dst[1] = (bf16_t)f4.y;
        dst[2] = (bf16_t)f4.z; dst[3] = (bf16_t)f4.w;
    }
    __syncthreads();

    const int nl    = lane & 15;          // N within tile / A row
    const int half  = lane >> 4;          // 0 or 1
    const int koffA = half * 16;          // A frag K sub-offset (halves)
    const int koffB = half * 16;          // B frag K sub-offset (halves)

    float sacc[8];
#pragma unroll
    for (int v = 0; v < 8; ++v) sacc[v] = 0.f;

    for (int ii = 0; ii < 4; ++ii) {
        const int n0 = (wave * 8 + 2 * ii) * 16 + nl;     // N-tile pair
        const int n1 = n0 + 16;
        const bf16_t* bcol0 = W1t + (size_t)n0 * F_;      // contiguous in k
        const bf16_t* bcol1 = W1t + (size_t)n1 * F_;
        v8f c0 = {};
        v8f c1 = {};
#pragma unroll 4
        for (int kb = 0; kb < F_; kb += 32) {
            union { v16bf v; v8bf h[2]; } a;
            const bf16_t* ap = &As[nl * LDS_STRIDE + kb + koffA];
            a.h[0] = *reinterpret_cast<const v8bf*>(ap);       // K = +0..7
            a.h[1] = *reinterpret_cast<const v8bf*>(ap + 8);   // K = +8..15
            const v16bf bb0 = *reinterpret_cast<const v16bf*>(bcol0 + kb + koffB);
            const v16bf bb1 = *reinterpret_cast<const v16bf*>(bcol1 + kb + koffB);
            c0 = __builtin_amdgcn_wmma_f32_16x16x32_bf16(
                     false, a.v, false, bb0, (short)0, c0, false, false);
            c1 = __builtin_amdgcn_wmma_f32_16x16x32_bf16(
                     false, a.v, false, bb1, (short)0, c1, false, false);
        }
        // post: C lane layout -> M = v + 8*half, N = nl (n uniform over v)
        const float cbn0 = cb[b * D_ + n0];
        const float wv0  = Wv[n0];
        const float cbn1 = cb[b * D_ + n1];
        const float wv1  = Wv[n1];
#pragma unroll
        for (int v = 0; v < 8; ++v) {
            sacc[v] = fmaf(wv0, fast_tanh(c0[v] + cbn0), sacc[v]);
            sacc[v] = fmaf(wv1, fast_tanh(c1[v] + cbn1), sacc[v]);
        }
    }

    // reduce over the 16 N-lanes of each half-wave (wave32 shuffles)
#pragma unroll
    for (int m = 1; m < 16; m <<= 1) {
#pragma unroll
        for (int v = 0; v < 8; ++v)
            sacc[v] += __shfl_xor(sacc[v], m, 32);
    }
    if (nl == 0) {
#pragma unroll
        for (int v = 0; v < 8; ++v)
            partials[wave][half * 8 + v] = sacc[v];
    }
    __syncthreads();
    if (tid < 16) {
        float s = bv[0];
#pragma unroll
        for (int w = 0; w < 8; ++w) s += partials[w][tid];
        scores[rowsBase + tid] = s;
    }
}

// ---------------------------------------------------------------------------
// Kernel C: softmax over T per batch. One block per batch, 8 elems/thread.
// ---------------------------------------------------------------------------
__global__ __launch_bounds__(256)
void softmax_kernel(const float* __restrict__ scores, float* __restrict__ attn) {
    __shared__ float red[256];
    const int b = blockIdx.x, tid = threadIdx.x;
    const float* s = scores + b * T_;
    float x[8];
    float m = -INFINITY;
#pragma unroll
    for (int j = 0; j < 8; ++j) { x[j] = s[tid + j * 256]; m = fmaxf(m, x[j]); }
    red[tid] = m; __syncthreads();
    for (int o = 128; o > 0; o >>= 1) {
        if (tid < o) red[tid] = fmaxf(red[tid], red[tid + o]);
        __syncthreads();
    }
    const float gmax = red[0]; __syncthreads();
    float sum = 0.f;
#pragma unroll
    for (int j = 0; j < 8; ++j) { x[j] = __expf(x[j] - gmax); sum += x[j]; }
    red[tid] = sum; __syncthreads();
    for (int o = 128; o > 0; o >>= 1) {
        if (tid < o) red[tid] += red[tid + o];
        __syncthreads();
    }
    const float inv = 1.f / red[0];
#pragma unroll
    for (int j = 0; j < 8; ++j) attn[b * T_ + tid + j * 256] = x[j] * inv;
}

// ---------------------------------------------------------------------------
// Kernel D: context[b,f] = sum_t attn[b,t] * features[b,t,f]
// ---------------------------------------------------------------------------
__global__ __launch_bounds__(256)
void context_kernel(const float* __restrict__ features,
                    const float* __restrict__ attn,
                    float* __restrict__ ctx) {
    const int b = blockIdx.x >> 2;
    const int f = (blockIdx.x & 3) * 256 + threadIdx.x;
    const float* fb = features + (size_t)b * T_ * F_ + f;
    const float* wb = attn + b * T_;
    float acc = 0.f;
    for (int t = 0; t < T_; t += 4) {
        acc = fmaf(wb[t + 0], fb[(size_t)(t + 0) * F_], acc);
        acc = fmaf(wb[t + 1], fb[(size_t)(t + 1) * F_], acc);
        acc = fmaf(wb[t + 2], fb[(size_t)(t + 2) * F_], acc);
        acc = fmaf(wb[t + 3], fb[(size_t)(t + 3) * F_], acc);
    }
    ctx[b * F_ + f] = acc;
}

// ---------------------------------------------------------------------------
extern "C" void kernel_launch(void* const* d_in, const int* in_sizes, int n_in,
                              void* d_out, int out_size, void* d_ws, size_t ws_size,
                              hipStream_t stream) {
    const float* hidden   = (const float*)d_in[0];
    const float* features = (const float*)d_in[1];
    const float* W1       = (const float*)d_in[2];
    const float* b1       = (const float*)d_in[3];
    const float* W2       = (const float*)d_in[4];
    const float* b2       = (const float*)d_in[5];
    const float* Wv       = (const float*)d_in[6];
    const float* bv       = (const float*)d_in[7];

    char* ws = (char*)d_ws;                       // ~2.4 MB total
    float*  cb     = (float*)(ws);                                   // B*D
    float*  scores = (float*)(ws + (size_t)B_ * D_ * 4);             // B*T
    bf16_t* W1t    = (bf16_t*)(ws + (size_t)B_ * D_ * 4
                                  + (size_t)B_ * T_ * 4);            // F*D bf16

    float* ctx  = (float*)d_out;                  // context_vector [B,F]
    float* attn = (float*)d_out + B_ * F_;        // attention_weights [B,T,1]

    prep_bias_kernel<<<B_, 256, 0, stream>>>(hidden, W2, b1, b2, cb);
    transpose_w1_kernel<<<1024, 256, 0, stream>>>(W1, W1t);
    attn_score_kernel<<<(B_ * T_) / 16, 256, 0, stream>>>(features, W1t, cb, Wv, bv, scores);
    softmax_kernel<<<B_, 256, 0, stream>>>(scores, attn);
    context_kernel<<<B_ * 4, 256, 0, stream>>>(features, attn, ctx);
}
